// HONAM_44521630990690
// MI455X (gfx1250) — compile-verified, matrix operand order checked by reading, not run on hardware
//
#include <hip/hip_runtime.h>
#include <hip/hip_bf16.h>

// ---------------------------------------------------------------------------
// HONAM on MI455X (gfx1250, wave32, WMMA).
//
// Per-feature MLP 1->32->64->32 (ReLU) over B=8192, F=256, then order-2
// Newton power-sum combine and 64->1 readout.
//
// Layer2/3 use v_wmma_f32_16x16x32_bf16. Weights are pre-swizzled into the
// WMMA B-register layout (bf16) by a prep kernel so each B operand is two
// global_load_b128 per lane. The layer2->layer3 activation repack goes
// through a per-wave LDS buffer laid out so the consumer side is plain
// ds_load_b128. Power sums accumulate in WMMA C-register layout and are
// reduced across feature groups with global f32 atomics.
// ---------------------------------------------------------------------------

typedef __attribute__((ext_vector_type(16))) __bf16 v16bf;
typedef __attribute__((ext_vector_type(8)))  float  v8f;

#define B_TOT     8192
#define F_TOT     256
#define H1        32
#define H2        64
#define H3        32
#define ROW_TILES (B_TOT / 16)       // 512
#define GROUPS    16                 // feature groups
#define FPG       (F_TOT / GROUPS)   // 16 features per group
#define WAVES     8                  // waves per block
#define NBLOCKS   ((ROW_TILES * GROUPS) / WAVES) // 1024

// ---------------------------------------------------------------------------
// Prep: convert W2/W3 (fp32, row-major [k][n]) into bf16 WMMA-B swizzled
// layout. Inverse of the consumer mapping: element e of lane L holds
// K = (e<8 ? e : e+8) + 8*(L>=16), N = ntile*16 + (L&15).
// W2sw: [F][4 ntiles][32 lanes][16 e]   (1 MB)
// W3sw: [F][2 kt][2 ntiles][32][16]     (1 MB)
// ---------------------------------------------------------------------------
__global__ void honam_prep_kernel(const float* __restrict__ W2,
                                  const float* __restrict__ W3,
                                  __bf16* __restrict__ W2sw,
                                  __bf16* __restrict__ W3sw)
{
    const int idx = blockIdx.x * blockDim.x + threadIdx.x;  // 0 .. F*2048-1
    if (idx >= F_TOT * 2048) return;
    const int f = idx >> 11;
    const int r = idx & 2047;
    {   // W2[f][k][n], k in 0..31, n in 0..63
        const int k = r >> 6, n = r & 63;
        const int nt   = n >> 4;
        const int lane = (n & 15) + 16 * ((k >> 3) & 1);
        const int e    = (k & 7) + 8 * (k >> 4);
        W2sw[(((size_t)f * 4 + nt) * 32 + lane) * 16 + e] = (__bf16)W2[idx];
    }
    {   // W3[f][k][n], k in 0..63, n in 0..31
        const int k = r >> 5, n = r & 31;
        const int kt = k >> 5, kk = k & 31;
        const int nt   = n >> 4;
        const int lane = (n & 15) + 16 * ((kk >> 3) & 1);
        const int e    = (kk & 7) + 8 * (kk >> 4);
        W3sw[((((size_t)f * 2 + kt) * 2 + nt) * 32 + lane) * 16 + e] = (__bf16)W3[idx];
    }
}

__global__ __launch_bounds__(WAVES * 32)
void honam_featnet_kernel(const float* __restrict__ x,
                          const float* __restrict__ W1, const float* __restrict__ b1,
                          const __bf16* __restrict__ W2sw, const float* __restrict__ b2,
                          const __bf16* __restrict__ W3sw, const float* __restrict__ b3,
                          float* __restrict__ p1g, float* __restrict__ p2g)
{
    // Small per-feature params (fp32), staged per block.
    __shared__ float w1s[H1], b1s[H1], b2s[H2], b3s[H3];
    // Per-wave h2 repack buffer in consumer (A-layout) order:
    // h2sw[wave][consumer_lane][kt] is one 32-byte v16bf operand slice.
    __shared__ v16bf h2sw[WAVES][32][2];    // 16 KB

    const int tid  = threadIdx.x;
    const int lane = tid & 31;
    const int wave = tid >> 5;

    const int unit     = blockIdx.x * WAVES + wave;  // 0..8191
    const int row_tile = unit % ROW_TILES;
    const int group    = unit / ROW_TILES;           // uniform within block
    const int b0       = row_tile * 16;

    const int hi   = (lane >= 16) ? 8 : 0;           // K-half / M-half selector
    const int mloc = lane & 15;

    v8f p1a0 = {0,0,0,0,0,0,0,0}, p1a1 = {0,0,0,0,0,0,0,0};
    v8f p2a0 = {0,0,0,0,0,0,0,0}, p2a1 = {0,0,0,0,0,0,0,0};

    for (int fi = 0; fi < FPG; ++fi) {
        const int f = group * FPG + fi;              // uniform across block

        const v16bf* w2p = (const v16bf*)W2sw + (size_t)f * 4 * 32;     // [nt][lane]
        const v16bf* w3p = (const v16bf*)W3sw + (size_t)f * 2 * 2 * 32; // [kt][nt][lane]

        // Prefetch next feature's swizzled weights while we compute this one.
        if (fi + 1 < FPG) {
            __builtin_prefetch((const v16bf*)W2sw + (size_t)(f + 1) * 4 * 32 + lane, 0, 1);
            __builtin_prefetch((const v16bf*)W3sw + (size_t)(f + 1) * 4 * 32 + lane, 0, 1);
        }

        // ---- stage per-feature scalar params ------------------------------
        __syncthreads();                 // prior-iteration readers done
        if (tid < H1) {
            w1s[tid] = W1[f * H1 + tid];
            b1s[tid] = b1[f * H1 + tid];
            b3s[tid] = b3[f * H3 + tid];
        }
        if (tid < H2) b2s[tid] = b2[f * H2 + tid];
        __syncthreads();

        // ---- layer 1: h1 = relu(x*W1 + b1), built directly in A layout ----
        const float xv = x[(size_t)(b0 + mloc) * F_TOT + f];
        v16bf a1;
#pragma unroll
        for (int e = 0; e < 16; ++e) {
            const int k = ((e < 8) ? e : e + 8) + hi;
            a1[e] = (__bf16)fmaxf(xv * w1s[k] + b1s[k], 0.0f);
        }

        // ---- layer 2: 16x64 = A(16x32) x W2(32x64), 4 N-tiles -------------
#pragma unroll
        for (int nt = 0; nt < 4; ++nt) {
            const v16bf bt = w2p[nt * 32 + lane];    // 2x global_load_b128
            v8f c = {0,0,0,0,0,0,0,0};
            c = __builtin_amdgcn_wmma_f32_16x16x32_bf16(
                    false, a1, false, bt, (short)0, c, false, false);
            const float bias = b2s[nt * 16 + mloc];
            // Scatter into consumer (A-layout) order.
            const int N    = nt * 16 + mloc;
            const int kt   = N >> 5;
            const int nn   = N & 31;
            const int Lrow = 16 * ((nn >> 3) & 1);   // + M below
            const int e    = (nn & 7) + 8 * ((nn >> 4) & 1);
#pragma unroll
            for (int r = 0; r < 8; ++r) {
                const float v = fmaxf(c[r] + bias, 0.0f);
                ((__bf16*)&h2sw[wave][Lrow + r + hi][kt])[e] = (__bf16)v;
            }
        }
        __syncthreads();                 // h2sw visible across lanes

        // ---- layer 3: A operands are two aligned 32B LDS loads ------------
        const v16bf a3lo = h2sw[wave][lane][0];      // ds_load_b128 x2
        const v16bf a3hi = h2sw[wave][lane][1];
#pragma unroll
        for (int nt = 0; nt < 2; ++nt) {
            const v16bf bt0 = w3p[(0 * 2 + nt) * 32 + lane];
            const v16bf bt1 = w3p[(1 * 2 + nt) * 32 + lane];
            v8f c = {0,0,0,0,0,0,0,0};
            c = __builtin_amdgcn_wmma_f32_16x16x32_bf16(
                    false, a3lo, false, bt0, (short)0, c, false, false);
            c = __builtin_amdgcn_wmma_f32_16x16x32_bf16(
                    false, a3hi, false, bt1, (short)0, c, false, false);
            const float bias = b3s[nt * 16 + mloc];
#pragma unroll
            for (int r = 0; r < 8; ++r) {
                const float t = fmaxf(c[r] + bias, 0.0f);
                if (nt == 0) { p1a0[r] += t; p2a0[r] += t * t; }
                else         { p1a1[r] += t; p2a1[r] += t * t; }
            }
        }
    }

    // ---- reduce partial power sums across feature groups -------------------
#pragma unroll
    for (int r = 0; r < 8; ++r) {
        const int brow = b0 + r + hi;
        atomicAdd(&p1g[brow * H3 + mloc],      p1a0[r]);
        atomicAdd(&p2g[brow * H3 + mloc],      p2a0[r]);
        atomicAdd(&p1g[brow * H3 + 16 + mloc], p1a1[r]);
        atomicAdd(&p2g[brow * H3 + 16 + mloc], p2a1[r]);
    }
}

__global__ void honam_combine_kernel(const float* __restrict__ p1g,
                                     const float* __restrict__ p2g,
                                     const float* __restrict__ Wout,
                                     const float* __restrict__ bout,
                                     float* __restrict__ out)
{
    const int b = blockIdx.x * blockDim.x + threadIdx.x;
    if (b >= B_TOT) return;
    float acc = bout[0];
#pragma unroll
    for (int h = 0; h < H3; ++h) {
        const float p1 = p1g[b * H3 + h];
        const float p2 = p2g[b * H3 + h];
        acc += p1 * Wout[h];                          // inters[1]
        acc += 0.5f * (p1 * p1 - p2) * Wout[H3 + h];  // inters[2]
    }
    out[b] = acc;
}

extern "C" void kernel_launch(void* const* d_in, const int* in_sizes, int n_in,
                              void* d_out, int out_size, void* d_ws, size_t ws_size,
                              hipStream_t stream)
{
    (void)in_sizes; (void)n_in; (void)out_size; (void)ws_size;

    const float* x    = (const float*)d_in[0];
    const float* W1   = (const float*)d_in[1];
    const float* b1   = (const float*)d_in[2];
    const float* W2   = (const float*)d_in[3];
    const float* b2   = (const float*)d_in[4];
    const float* W3   = (const float*)d_in[5];
    const float* b3   = (const float*)d_in[6];
    const float* Wout = (const float*)d_in[7];
    const float* bout = (const float*)d_in[8];
    float* out = (float*)d_out;

    // Workspace layout:
    //   [0, 1MB)   p1g  [B, H3] fp32
    //   [1, 2MB)   p2g  [B, H3] fp32
    //   [2, 3MB)   W2sw [F][4][32][16] bf16
    //   [3, 4MB)   W3sw [F][2][2][32][16] bf16
    char* ws = (char*)d_ws;
    float*  p1g  = (float*)ws;
    float*  p2g  = (float*)(ws + (size_t)B_TOT * H3 * sizeof(float));
    __bf16* W2sw = (__bf16*)(ws + 2 * (size_t)B_TOT * H3 * sizeof(float));
    __bf16* W3sw = W2sw + (size_t)F_TOT * H1 * H2;

    // Zero power-sum accumulators each call (graph-capture safe).
    hipMemsetAsync(d_ws, 0, 2 * (size_t)B_TOT * H3 * sizeof(float), stream);

    honam_prep_kernel<<<(F_TOT * 2048) / 256, 256, 0, stream>>>(W2, W3, W2sw, W3sw);

    honam_featnet_kernel<<<NBLOCKS, WAVES * 32, 0, stream>>>(
        x, W1, b1, W2sw, b2, W3sw, b3, p1g, p2g);

    honam_combine_kernel<<<(B_TOT + 255) / 256, 256, 0, stream>>>(
        p1g, p2g, Wout, bout, out);
}
